// graph_detector_4612794876552
// MI455X (gfx1250) — matrix operand order repaired and checked by reading, not compiled, not run on hardware
//
#include <hip/hip_runtime.h>
#include <hip/hip_bf16.h>
#include <math.h>

// ---------------------------------------------------------------------------
// SuperGlue-style matcher for MI455X (gfx1250, wave32).
// - GEMM inputs are bf16 in global memory; math is v_wmma_f32_16x16x32_bf16.
// - Fully k-contiguous GEMMs run a double-buffered DMA pipeline: the Tensor
//   Data Mover (tensor_load_to_lds, TENSORcnt) fetches tile k+1 (with
//   hardware LDS row padding) while WMMA consumes tile k. Falls back to
//   global_load_async_to_lds_b128 chains if the TDM builtin is unavailable.
// - Other layouts use branch-free full-tile vector staging; clamped code only
//   for genuinely partial tiles (encoder layer 0).
// ---------------------------------------------------------------------------

typedef __bf16 bf16_t;
typedef bf16_t v16bf __attribute__((ext_vector_type(16)));
typedef float  v8f   __attribute__((ext_vector_type(8)));
typedef unsigned int v4u __attribute__((ext_vector_type(4)));
typedef int v8i __attribute__((ext_vector_type(8)));
typedef int v4i __attribute__((ext_vector_type(4)));

#define DEVINL __device__ __forceinline__

#if __has_builtin(__builtin_amdgcn_tensor_load_to_lds)
#define HAVE_TDM 1
#endif

DEVINL unsigned short f2bf(float f) {
  union { float f; unsigned u; } x; x.f = f;
  unsigned r = x.u + 0x7fffu + ((x.u >> 16) & 1u);   // round-to-nearest-even
  return (unsigned short)(r >> 16);
}

DEVINL void async_ld_b128(unsigned lds_off, const void* gaddr) {
  asm volatile("global_load_async_to_lds_b128 %0, %1, off"
               :: "v"(lds_off), "v"(gaddr) : "memory");
}
DEVINL void wait_async_le0() { asm volatile("s_wait_asynccnt 0x0" ::: "memory"); }
DEVINL void wait_async_le4() { asm volatile("s_wait_asynccnt 0x4" ::: "memory"); }

#ifdef HAVE_TDM
// One TDM descriptor: 2D tile (tile_k bf16 along contiguous dim0, tile_rows
// lines at row_stride elements), stored to LDS with 4-DWORD padding every
// 16 DWORDs (=> 80B LDS row pitch matching u16[rows][40] tiles).
DEVINL void tdm_load_2d(unsigned lds_byte_off, const void* gaddr,
                        unsigned tile_k, unsigned tile_rows,
                        unsigned long row_stride_elems) {
  unsigned long ga = (unsigned long)(uintptr_t)gaddr;
  v4u g0;
  g0[0] = 1u;                                        // count=1 (user D#)
  g0[1] = lds_byte_off;                              // lds_addr
  g0[2] = (unsigned)(ga & 0xffffffffu);              // global_addr[31:0]
  g0[3] = (unsigned)((ga >> 32) & 0x01ffffffu)       // global_addr[56:32]
        | (2u << 30);                                // type=2 (image)
  v8i g1;
  g1[0] = (int)((1u << 16)      // data_size = 2 bytes (bf16)
              | (1u << 20)      // pad_enable
              | (3u << 22)      // pad_interval: 16 DWORDs of data
              | (3u << 25));    // pad_amount: 4 DWORDs of pad
  g1[1] = (int)0xffff0000u;                 // tensor_dim0[15:0] (huge, no OOB)
  g1[2] = (int)(0x7fffu | (0xffffu << 16)); // tensor_dim0 hi, tensor_dim1 lo
  g1[3] = (int)(0x7fffu | (tile_k << 16));  // tensor_dim1 hi, tile_dim0
  g1[4] = (int)tile_rows;                   // tile_dim1 (tile_dim2 = 0)
  g1[5] = (int)(unsigned)(row_stride_elems & 0xffffffffu); // dim0_stride lo
  g1[6] = (int)(unsigned)((row_stride_elems >> 32) & 0xffffu); // stride hi
  g1[7] = 0;
  v4i gz4 = {0, 0, 0, 0};
  v8i gz8 = {0, 0, 0, 0, 0, 0, 0, 0};
  __builtin_amdgcn_tensor_load_to_lds(g0, g1, gz4, gz4, gz8, 0);
}
DEVINL void wait_tensor_le0() { __builtin_amdgcn_s_wait_tensorcnt(0); }
DEVINL void wait_tensor_le2() { __builtin_amdgcn_s_wait_tensorcnt(2); }
#endif

#define GF_BIAS   1
#define GF_BNRELU 2
#define GF_RES    4
#define GF_OUTF   8
#define GF_OUTB   16

typedef unsigned short u16;

// C[m][n] = scale * sum_k A[m,k]*B[k,n] (+bias/bn/relu/residual)
// A: bf16 at A + z*abs_ + m*ars + k*acs ; B: bf16 at B + z*bbs + k*brs + n*bcs
// Outputs: fp32 Cf and/or bf16 Cb, row stride crs, batch stride cbs.
__global__ __launch_bounds__(128) void wmma_gemm_kernel(
    const u16* __restrict__ A, long ars, long acs, long abs_,
    const u16* __restrict__ Bm, long brs, long bcs, long bbs,
    float* __restrict__ Cf, u16* __restrict__ Cb, long crs, long cbs,
    const float* __restrict__ bias,
    const float* __restrict__ bng, const float* __restrict__ bnb,
    const float* __restrict__ bnm, const float* __restrict__ bnv,
    const float* __restrict__ Res, long resbs,
    int M, int N, int K, float scale, int flags)
{
  __shared__ u16 As[2][64][40];   // +8 ushort pad: conflict-free ds_load_b128
  __shared__ u16 Bs[2][64][40];

  const int tid  = threadIdx.x;
  const int lane = tid & 31;
  const int wave = tid >> 5;
  const int lhi  = lane >> 4;
  const int llo  = lane & 15;
  const int m0   = blockIdx.y * 64;
  const int n0   = blockIdx.x * 64;
  const long z   = blockIdx.z;

  const u16* Ab = A  + z * abs_;
  const u16* Bb = Bm + z * bbs;

  const bool fullM = (m0 + 64 <= M);
  const bool fullN = (n0 + 64 <= N);
  const bool aAsyncOK = (acs == 1) && fullM && ((ars & 7) == 0) &&
                        (((unsigned long)(uintptr_t)Ab & 15) == 0);
  const bool bAsyncOK = (brs == 1) && fullN && ((bcs & 7) == 0) &&
                        (((unsigned long)(uintptr_t)Bb & 15) == 0);

  v8f acc[4];
  #pragma unroll
  for (int g = 0; g < 4; ++g)
    #pragma unroll
    for (int r = 0; r < 8; ++r) acc[g][r] = 0.0f;

  const int kA0 = lhi ? 16 : 0;  // A frag K groups {0-15} / {16-31}
  const int kB0 = lhi ? 16 : 0;  // B frag K groups {0-15} / {16-31}

  auto compute_tile = [&](int p) {
    union Frag { v16bf v; uint4 q[2]; } af, bfr;
    const u16* arow = &As[p][wave * 16 + llo][0];
    af.q[0] = *(const uint4*)(arow + kA0);
    af.q[1] = *(const uint4*)(arow + kA0 + 8);
    #pragma unroll
    for (int g = 0; g < 4; ++g) {
      const u16* brow = &Bs[p][g * 16 + llo][0];
      bfr.q[0] = *(const uint4*)(brow + kB0);
      bfr.q[1] = *(const uint4*)(brow + kB0 + 8);
      acc[g] = __builtin_amdgcn_wmma_f32_16x16x32_bf16(
          false, af.v, false, bfr.v, (short)0, acc[g], false, false);
    }
  };

  if (aAsyncOK && bAsyncOK && ((K & 31) == 0)) {
    // ------------- double-buffered DMA pipeline (TDM preferred) -------------
    const int nk = K >> 5;
    int p = 0;
#ifdef HAVE_TDM
    auto stage = [&](int k0, int pp) {
      tdm_load_2d((unsigned)(uintptr_t)&As[pp][0][0],
                  Ab + (long)m0 * ars + k0, 32u, 64u, (unsigned long)ars);
      tdm_load_2d((unsigned)(uintptr_t)&Bs[pp][0][0],
                  Bb + (long)n0 * bcs + k0, 32u, 64u, (unsigned long)bcs);
    };
    if (wave == 0) stage(0, 0);
    for (int kt = 0; kt < nk; ++kt) {
      if (wave == 0) {
        if (kt + 1 < nk) { stage((kt + 1) * 32, p ^ 1); wait_tensor_le2(); }
        else             { wait_tensor_le0(); }
      }
      __syncthreads();
      compute_tile(p);
      __syncthreads();
      p ^= 1;
    }
#else
    auto stage = [&](int k0, int pp) {
      #pragma unroll
      for (int i = 0; i < 2; ++i) {
        int idx = tid + i * 128;
        int r = idx >> 2, c8 = (idx & 3) * 8;
        async_ld_b128((unsigned)(uintptr_t)&As[pp][r][c8],
                      Ab + (long)(m0 + r) * ars + (k0 + c8));
        async_ld_b128((unsigned)(uintptr_t)&Bs[pp][r][c8],
                      Bb + (long)(n0 + r) * bcs + (k0 + c8));
      }
    };
    stage(0, 0);
    for (int kt = 0; kt < nk; ++kt) {
      if (kt + 1 < nk) { stage((kt + 1) * 32, p ^ 1); wait_async_le4(); }
      else             { wait_async_le0(); }
      __syncthreads();
      compute_tile(p);
      __syncthreads();
      p ^= 1;
    }
#endif
  } else {
    // ---------------- generic single-buffer loop ----------------
    for (int k0 = 0; k0 < K; k0 += 32) {
      const bool fullK = (k0 + 32 <= K);

      // ---- stage A tile: As[0][m_local][k_local] ----
      if (aAsyncOK && fullK) {
        #pragma unroll
        for (int i = 0; i < 2; ++i) {
          int idx = tid + i * 128;
          int r = idx >> 2, c8 = (idx & 3) * 8;
          async_ld_b128((unsigned)(uintptr_t)&As[0][r][c8],
                        Ab + (long)(m0 + r) * ars + (k0 + c8));
        }
      } else if (fullM && fullK) {
        if (ars == 1 && ((acs & 3) == 0) &&
            (((unsigned long)(uintptr_t)Ab & 7) == 0)) {
          // vectorized, coalesced along m (4 bf16 per load)
          #pragma unroll
          for (int i = 0; i < 4; ++i) {
            int idx = tid + i * 128;
            int c = idx >> 4, r4 = (idx & 15) * 4;
            union { uint2 u; u16 h[4]; } t;
            t.u = *(const uint2*)(Ab + (m0 + r4) + (long)(k0 + c) * acs);
            As[0][r4 + 0][c] = t.h[0];
            As[0][r4 + 1][c] = t.h[1];
            As[0][r4 + 2][c] = t.h[2];
            As[0][r4 + 3][c] = t.h[3];
          }
        } else if (ars == 1) {
          #pragma unroll
          for (int i = 0; i < 16; ++i) {
            int idx = tid + i * 128;
            int c = idx >> 6, r = idx & 63;
            As[0][r][c] = Ab[(m0 + r) + (long)(k0 + c) * acs];
          }
        } else {
          #pragma unroll
          for (int i = 0; i < 16; ++i) {
            int idx = tid + i * 128;
            int r = idx >> 5, c = idx & 31;
            As[0][r][c] = Ab[(long)(m0 + r) * ars + (long)(k0 + c) * acs];
          }
        }
      } else {
        #pragma unroll 1
        for (int i = 0; i < 16; ++i) {
          int idx = tid + i * 128;
          int r = idx >> 5, c = idx & 31;
          int m = m0 + r, k = k0 + c;
          long mc = (m < M) ? m : (M - 1);
          long kc = (k < K) ? k : (K - 1);
          u16 v = Ab[mc * ars + kc * acs];
          As[0][r][c] = (m < M && k < K) ? v : (u16)0;
        }
      }

      // ---- stage B tile transposed: Bs[0][n_local][k_local] ----
      if (bAsyncOK && fullK) {
        #pragma unroll
        for (int i = 0; i < 2; ++i) {
          int idx = tid + i * 128;
          int r = idx >> 2, c8 = (idx & 3) * 8;
          async_ld_b128((unsigned)(uintptr_t)&Bs[0][r][c8],
                        Bb + (long)(n0 + r) * bcs + (k0 + c8));
        }
      } else if (fullN && fullK) {
        if (bcs == 1 && ((brs & 3) == 0) &&
            (((unsigned long)(uintptr_t)Bb & 7) == 0)) {
          // vectorized, coalesced along n
          #pragma unroll
          for (int i = 0; i < 4; ++i) {
            int idx = tid + i * 128;
            int c = idx >> 4, r4 = (idx & 15) * 4;
            union { uint2 u; u16 h[4]; } t;
            t.u = *(const uint2*)(Bb + (long)(k0 + c) * brs + (n0 + r4));
            Bs[0][r4 + 0][c] = t.h[0];
            Bs[0][r4 + 1][c] = t.h[1];
            Bs[0][r4 + 2][c] = t.h[2];
            Bs[0][r4 + 3][c] = t.h[3];
          }
        } else if (bcs == 1) {
          #pragma unroll
          for (int i = 0; i < 16; ++i) {
            int idx = tid + i * 128;
            int c = idx >> 6, r = idx & 63;
            Bs[0][r][c] = Bb[(long)(k0 + c) * brs + (n0 + r)];
          }
        } else {
          #pragma unroll
          for (int i = 0; i < 16; ++i) {
            int idx = tid + i * 128;
            int r = idx >> 5, c = idx & 31;
            Bs[0][r][c] = Bb[(long)(k0 + c) * brs + (long)(n0 + r) * bcs];
          }
        }
      } else {
        #pragma unroll 1
        for (int i = 0; i < 16; ++i) {
          int idx = tid + i * 128;
          int r = idx >> 5, c = idx & 31;
          int n = n0 + r, k = k0 + c;
          long nc = (n < N) ? n : (N - 1);
          long kc = (k < K) ? k : (K - 1);
          u16 v = Bb[kc * brs + nc * bcs];
          Bs[0][r][c] = (n < N && k < K) ? v : (u16)0;
        }
      }

      wait_async_le0();
      __syncthreads();
      compute_tile(0);
      __syncthreads();
    }
  }

  // ---------------- epilogue ----------------
  float* Cfb = Cf + z * cbs;
  u16* Cbb = Cb + z * cbs;
  const float* Rb = Res + z * resbs;
  #pragma unroll
  for (int g = 0; g < 4; ++g) {
    #pragma unroll
    for (int r = 0; r < 8; ++r) {
      int m = m0 + wave * 16 + lhi * 8 + r;
      int n = n0 + g * 16 + llo;
      if (m < M && n < N) {
        float x = acc[g][r] * scale;
        if (flags & GF_BIAS) x += bias[m];
        if (flags & GF_BNRELU) {
          float s = bng[m] * rsqrtf(bnv[m] + 1e-5f);
          x = (x - bnm[m]) * s + bnb[m];
          x = fmaxf(x, 0.0f);
        }
        if (flags & GF_RES) x += Rb[(long)m * crs + n];
        if (flags & GF_OUTF) Cfb[(long)m * crs + n] = x;
        if (flags & GF_OUTB) Cbb[(long)m * crs + n] = f2bf(x);
      }
    }
  }
}

// ---------------- elementwise / reduction kernels ----------------

__global__ void f32_to_bf16_kernel(const float* __restrict__ s,
                                   u16* __restrict__ d, long n) {
  long t = (long)blockIdx.x * blockDim.x + threadIdx.x;
  if (t < n) d[t] = f2bf(s[t]);
}

__global__ void enc_input_kernel(const float* __restrict__ kpts,
                                 const float* __restrict__ kscores,
                                 const int* __restrict__ image_size,
                                 u16* __restrict__ out, int N, int nb) {
  long t = (long)blockIdx.x * blockDim.x + threadIdx.x;
  long total = (long)nb * N;
  if (t >= total) return;
  int b = (int)(t / N), n = (int)(t % N);
  float W = (float)image_size[1], H = (float)image_size[0];
  float scale = fmaxf(W, H) * 0.7f;
  float kx = kpts[((long)b * N + n) * 2 + 0];
  float ky = kpts[((long)b * N + n) * 2 + 1];
  u16* ob = out + (long)b * 3 * N;
  ob[0 * N + n] = f2bf((kx - W * 0.5f) / scale);
  ob[1 * N + n] = f2bf((ky - H * 0.5f) / scale);
  ob[2 * N + n] = f2bf(kscores[(long)b * N + n]);
}

__global__ void copy_rows_u16_kernel(const u16* __restrict__ src, long sbs,
                                     u16* __restrict__ dst, long dbs,
                                     long cnt, int nb) {
  long t = (long)blockIdx.x * blockDim.x + threadIdx.x;
  long total = (long)nb * cnt;
  if (t >= total) return;
  int b = (int)(t / cnt);
  long r = t % cnt;
  dst[(long)b * dbs + r] = src[(long)b * sbs + r];
}

__global__ void fill_kernel(float* __restrict__ p, float v, long n) {
  long t = (long)blockIdx.x * blockDim.x + threadIdx.x;
  if (t < n) p[t] = v;
}

// softmax over fp32 rows -> bf16 output rows (one wave per row)
__global__ void softmax_rows_bf16_kernel(const float* __restrict__ S,
                                         u16* __restrict__ O,
                                         int len, long rows) {
  long w = ((long)blockIdx.x * blockDim.x + threadIdx.x) >> 5;
  if (w >= rows) return;
  int lane = threadIdx.x & 31;
  const float* row = S + w * (long)len;
  u16* orow = O + w * (long)len;
  float mx = -3.402823466e38f;
  for (int j = lane; j < len; j += 32) mx = fmaxf(mx, row[j]);
  for (int o = 16; o; o >>= 1) mx = fmaxf(mx, __shfl_xor(mx, o, 32));
  float s = 0.0f;
  for (int j = lane; j < len; j += 32) s += __expf(row[j] - mx);
  for (int o = 16; o; o >>= 1) s += __shfl_xor(s, o, 32);
  float lse = mx + __logf(s);
  for (int j = lane; j < len; j += 32) orow[j] = f2bf(__expf(row[j] - lse));
}

__global__ void build_couplings_kernel(const float* __restrict__ sc,
                                       const float* __restrict__ alpha,
                                       float* __restrict__ cpl,
                                       int n0, int n1, int nb) {
  long t = (long)blockIdx.x * blockDim.x + threadIdx.x;
  long per = (long)(n0 + 1) * (n1 + 1);
  long total = per * nb;
  if (t >= total) return;
  int b = (int)(t / per);
  long r = t % per;
  int i = (int)(r / (n1 + 1)), j = (int)(r % (n1 + 1));
  float v = (i < n0 && j < n1) ? sc[((long)b * n0 + i) * n1 + j] : alpha[0];
  cpl[t] = v;
}

__global__ void sink_row_kernel(const float* __restrict__ cpl,
                                const float* __restrict__ vv,
                                float* __restrict__ uu,
                                int n, int nb, float norm, float logns) {
  long w = ((long)blockIdx.x * blockDim.x + threadIdx.x) >> 5;
  long rows = (long)nb * n;
  if (w >= rows) return;
  int lane = threadIdx.x & 31;
  int b = (int)(w / n), i = (int)(w % n);
  const float* row = cpl + ((long)b * n + i) * n;
  const float* vb = vv + (long)b * n;
  float mx = -3.402823466e38f;
  for (int j = lane; j < n; j += 32) mx = fmaxf(mx, row[j] + vb[j]);
  for (int o = 16; o; o >>= 1) mx = fmaxf(mx, __shfl_xor(mx, o, 32));
  float s = 0.0f;
  for (int j = lane; j < n; j += 32) s += __expf(row[j] + vb[j] - mx);
  for (int o = 16; o; o >>= 1) s += __shfl_xor(s, o, 32);
  if (lane == 0) {
    float lmu = (i < n - 1) ? norm : (logns + norm);
    uu[(long)b * n + i] = lmu - (mx + __logf(s));
  }
}

__global__ void sink_col_kernel(const float* __restrict__ cpl,
                                const float* __restrict__ uu,
                                float* __restrict__ vv,
                                int n, int nb, float norm, float logms) {
  long t = (long)blockIdx.x * blockDim.x + threadIdx.x;
  long total = (long)nb * n;
  if (t >= total) return;
  int b = (int)(t / n), j = (int)(t % n);
  const float* base = cpl + (long)b * n * n;
  const float* ub = uu + (long)b * n;
  float mx = -3.402823466e38f;
  for (int i = 0; i < n; ++i) mx = fmaxf(mx, base[(long)i * n + j] + ub[i]);
  float s = 0.0f;
  for (int i = 0; i < n; ++i) s += __expf(base[(long)i * n + j] + ub[i] - mx);
  float lnu = (j < n - 1) ? norm : (logms + norm);
  vv[(long)b * n + j] = lnu - (mx + __logf(s));
}

__global__ void finalize_z_kernel(const float* __restrict__ cpl,
                                  const float* __restrict__ uu,
                                  const float* __restrict__ vv,
                                  float* __restrict__ Z,
                                  int n, int nb, float norm) {
  long t = (long)blockIdx.x * blockDim.x + threadIdx.x;
  long per = (long)n * n;
  long total = per * nb;
  if (t >= total) return;
  int b = (int)(t / per);
  long r = t % per;
  int i = (int)(r / n), j = (int)(r % n);
  Z[t] = cpl[t] + uu[(long)b * n + i] + vv[(long)b * n + j] - norm;
}

__global__ void argmax_row_kernel(const float* __restrict__ Z,
                                  float* __restrict__ mx0, int* __restrict__ ix0,
                                  int n, int nz, int nb) {
  long w = ((long)blockIdx.x * blockDim.x + threadIdx.x) >> 5;
  long rows = (long)nb * n;
  if (w >= rows) return;
  int lane = threadIdx.x & 31;
  int b = (int)(w / n), i = (int)(w % n);
  const float* row = Z + ((long)b * nz + i) * nz;
  float mx = -3.402823466e38f; int idx = 0;
  for (int j = lane; j < n; j += 32) {
    float x = row[j];
    if (x > mx) { mx = x; idx = j; }
  }
  for (int o = 16; o; o >>= 1) {
    float om = __shfl_xor(mx, o, 32);
    int   oi = __shfl_xor(idx, o, 32);
    if (om > mx || (om == mx && oi < idx)) { mx = om; idx = oi; }
  }
  if (lane == 0) { mx0[(long)b * n + i] = mx; ix0[(long)b * n + i] = idx; }
}

__global__ void argmax_col_kernel(const float* __restrict__ Z,
                                  float* __restrict__ mx1, int* __restrict__ ix1,
                                  int n, int nz, int nb) {
  long t = (long)blockIdx.x * blockDim.x + threadIdx.x;
  long total = (long)nb * n;
  if (t >= total) return;
  int b = (int)(t / n), j = (int)(t % n);
  const float* base = Z + (long)b * nz * nz;
  float mx = -3.402823466e38f; int idx = 0;
  for (int i = 0; i < n; ++i) {
    float x = base[(long)i * nz + j];
    if (x > mx) { mx = x; idx = i; }
  }
  mx1[(long)b * n + j] = mx;
  ix1[(long)b * n + j] = idx;
}

__global__ void match0_kernel(const float* __restrict__ mx0,
                              const int* __restrict__ ix0,
                              const int* __restrict__ ix1,
                              int* __restrict__ matches0,
                              float* __restrict__ msc0, int n, int nb) {
  long t = (long)blockIdx.x * blockDim.x + threadIdx.x;
  long total = (long)nb * n;
  if (t >= total) return;
  int b = (int)(t / n), i = (int)(t % n);
  int j = ix0[(long)b * n + i];
  bool mutual = (ix1[(long)b * n + j] == i);
  float msc = mutual ? __expf(mx0[(long)b * n + i]) : 0.0f;
  matches0[(long)b * n + i] = (mutual && msc > 0.7f) ? j : -1;
  msc0[(long)b * n + i] = msc;
}

__global__ void match1_kernel(const int* __restrict__ ix0,
                              const int* __restrict__ ix1,
                              const int* __restrict__ matches0,
                              const float* __restrict__ msc0,
                              int* __restrict__ matches1,
                              float* __restrict__ msc1, int n, int nb) {
  long t = (long)blockIdx.x * blockDim.x + threadIdx.x;
  long total = (long)nb * n;
  if (t >= total) return;
  int b = (int)(t / n), j = (int)(t % n);
  int i = ix1[(long)b * n + j];
  bool mutual = (ix0[(long)b * n + i] == j);
  float m = mutual ? msc0[(long)b * n + i] : 0.0f;
  matches1[(long)b * n + j] = (mutual && matches0[(long)b * n + i] >= 0) ? i : -1;
  msc1[(long)b * n + j] = m;
}

// ---------------- host orchestration ----------------

static inline void launch_gemm(hipStream_t stream,
    const u16* A, long ars, long acs, long abs_,
    const u16* Bp, long brs, long bcs, long bbs,
    float* Cf, u16* Cb, long crs, long cbs,
    int M, int N, int K, int nb, float scale, int flags,
    const float* bias = nullptr,
    const float* bng = nullptr, const float* bnb = nullptr,
    const float* bnm = nullptr, const float* bnv = nullptr,
    const float* res = nullptr, long resbs = 0)
{
  dim3 grid((unsigned)((N + 63) / 64), (unsigned)((M + 63) / 64), (unsigned)nb);
  wmma_gemm_kernel<<<grid, dim3(128), 0, stream>>>(
      A, ars, acs, abs_, Bp, brs, bcs, bbs, Cf, Cb, crs, cbs,
      bias, bng, bnb, bnm, bnv, res, resbs, M, N, K, scale, flags);
}

extern "C" void kernel_launch(void* const* d_in, const int* in_sizes, int n_in,
                              void* d_out, int out_size, void* d_ws, size_t ws_size,
                              hipStream_t stream) {
  (void)in_sizes; (void)n_in; (void)out_size; (void)ws_size;
  const int BATCH = 2, N = 1024, D = 256, HEADS = 4;
  const int NZ = N + 1;
  const long DN = (long)D * N;

  auto P = [&](int i) { return (const float*)d_in[i]; };
  const float* desc0  = P(0);
  const float* desc1  = P(1);
  const float* kpts0  = P(2);
  const float* kpts1  = P(3);
  const float* scr0   = P(4);
  const float* scr1   = P(5);
  const int*   img_sz = (const int*)d_in[6];
  const float* alpha  = P(7);   // bin_score
  const int ENC_BASE = 8, FP_B = 34, FP_W = 35, GNN_BASE = 36;

  const int TPB = 256;
  auto blocks = [&](long n) { return (unsigned)((n + TPB - 1) / TPB); };

  // ---- fp32 workspace ----
  float* wsf = (float*)d_ws;
  long foff = 0;
  auto falloc = [&](long n) { float* p = wsf + foff; foff += (n + 3) & ~3L; return p; };
  float* d0f  = falloc(BATCH * DN);
  float* d1f  = falloc(BATCH * DN);
  float* bigf = falloc((long)BATCH * HEADS * N * N); // attn scores; later sc+cpl
  float* uu   = falloc((long)BATCH * NZ);
  float* vv   = falloc((long)BATCH * NZ);
  float* mx0  = falloc((long)BATCH * N);
  int*   ix0  = (int*)falloc((long)BATCH * N);
  float* mx1  = falloc((long)BATCH * N);
  int*   ix1  = (int*)falloc((long)BATCH * N);

  // ---- bf16 workspace (16B aligned) ----
  u16* usp = (u16*)(wsf + foff);
  long uoff = 0;
  auto ualloc = [&](long n) { u16* p = usp + uoff; uoff += (n + 7) & ~7L; return p; };
  auto cvt = [&](int idx, long n) {
    u16* d = ualloc(n);
    f32_to_bf16_kernel<<<blocks(n), TPB, 0, stream>>>(P(idx), d, n);
    return d;
  };

  // weight shadows (bf16), converted once per call
  const int ench[6] = {3, 32, 64, 128, 256, 256};
  u16* encw[5];
  {
    int base = ENC_BASE;
    for (int l = 0; l < 5; ++l) {
      bool has_bn = (l < 4);
      int widx = has_bn ? base + 5 : base + 1;
      encw[l] = cvt(widx, (long)ench[l + 1] * ench[l]);
      base += has_bn ? 6 : 2;
    }
  }
  u16* fpw = cvt(FP_W, (long)D * D);
  u16 *gqw[18], *gkw[18], *gvw[18], *gmw[18], *gc1w[18], *gc2w[18];
  for (int l = 0; l < 18; ++l) {
    int gb = GNN_BASE + 16 * l;
    gkw[l]  = cvt(gb + 1,  (long)D * D);
    gmw[l]  = cvt(gb + 3,  (long)D * D);
    gc1w[l] = cvt(gb + 9,  (long)2 * D * 2 * D);
    gc2w[l] = cvt(gb + 11, (long)D * 2 * D);
    gqw[l]  = cvt(gb + 13, (long)D * D);
    gvw[l]  = cvt(gb + 15, (long)D * D);
  }

  // bf16 activation buffers
  u16* encinb = ualloc((long)BATCH * 3 * N);
  u16* d0b  = ualloc(BATCH * DN);
  u16* d1b  = ualloc(BATCH * DN);
  u16* hAb  = ualloc(BATCH * DN);
  u16* hBb  = ualloc(BATCH * DN);
  u16* qb16 = ualloc(BATCH * DN);
  u16* kb16 = ualloc(BATCH * DN);
  u16* vb16 = ualloc(BATCH * DN);
  u16* ob16 = ualloc(BATCH * DN);
  u16* attnb = ualloc((long)BATCH * HEADS * N * N);
  u16* c0b  = ualloc((long)BATCH * 2 * DN);
  u16* c1b_ = ualloc((long)BATCH * 2 * DN);
  u16* hidb = ualloc((long)BATCH * 2 * DN);
  u16* f0b  = ualloc(BATCH * DN);
  u16* f1b  = ualloc(BATCH * DN);

  // ---- outputs ----
  float* Z = (float*)d_out;
  long zlen = (long)BATCH * NZ * NZ;
  int*   matches0 = (int*)(Z + zlen);
  int*   matches1 = matches0 + (long)BATCH * N;
  float* msc0     = (float*)(matches1 + (long)BATCH * N);
  float* msc1     = msc0 + (long)BATCH * N;

  // =====================  keypoint encoder  =====================
  for (int t = 0; t < 2; ++t) {
    const float* kp = t ? kpts1 : kpts0;
    const float* sc = t ? scr1 : scr0;
    const float* dd = t ? desc1 : desc0;
    float* doutf = t ? d1f : d0f;
    u16*   doutb = t ? d1b : d0b;
    enc_input_kernel<<<blocks((long)BATCH * N), TPB, 0, stream>>>(
        kp, sc, img_sz, encinb, N, BATCH);
    const u16* x = encinb;
    int base = ENC_BASE;
    u16* ping = hAb, * pong = hBb;
    for (int l = 0; l < 5; ++l) {
      int cin = ench[l], cout = ench[l + 1];
      bool has_bn = (l < 4);
      const float *bnbp=nullptr,*bngp=nullptr,*bnmp=nullptr,*bnvp=nullptr,*cb;
      if (has_bn) { bnbp=P(base); bngp=P(base+1); bnmp=P(base+2); bnvp=P(base+3);
                    cb=P(base+4); base += 6; }
      else        { cb=P(base); base += 2; }
      bool last = (l == 4);
      int flags = GF_BIAS | (has_bn ? GF_BNRELU : 0)
                | (last ? (GF_RES | GF_OUTF | GF_OUTB) : GF_OUTB);
      launch_gemm(stream, encw[l], cin, 1, 0,
                  x, N, 1, (long)cin * N,
                  last ? doutf : nullptr, last ? doutb : ping,
                  N, last ? DN : (long)cout * N,
                  cout, N, cin, BATCH, 1.0f, flags,
                  cb, bngp, bnbp, bnmp, bnvp,
                  last ? dd : nullptr, DN);
      x = last ? doutb : ping;
      u16* tmp = ping; ping = pong; pong = tmp;
    }
  }

  // =====================  GNN layers  =====================
  for (int l = 0; l < 18; ++l) {
    int gb = GNN_BASE + 16 * l;
    const float *kb=P(gb+0),*mb=P(gb+2);
    const float *bnbp=P(gb+4),*bngp=P(gb+5),*bnmp=P(gb+6),*bnvp=P(gb+7);
    const float *c1b=P(gb+8),*c2b=P(gb+10),*qb=P(gb+12),*vb=P(gb+14);
    bool cross = (l & 1);

    for (int t = 0; t < 2; ++t) {
      const u16* x = t ? d1b : d0b;
      const u16* s = t ? (cross ? d0b : d1b) : (cross ? d1b : d0b);
      u16* cc = t ? c1b_ : c0b;
      // q,k,v projections (bf16 out)
      launch_gemm(stream, gqw[l], D, 1, 0, x, N, 1, DN, nullptr, qb16, N, DN,
                  D, N, D, BATCH, 1.0f, GF_BIAS | GF_OUTB, qb);
      launch_gemm(stream, gkw[l], D, 1, 0, s, N, 1, DN, nullptr, kb16, N, DN,
                  D, N, D, BATCH, 1.0f, GF_BIAS | GF_OUTB, kb);
      launch_gemm(stream, gvw[l], D, 1, 0, s, N, 1, DN, nullptr, vb16, N, DN,
                  D, N, D, BATCH, 1.0f, GF_BIAS | GF_OUTB, vb);
      // scores[h][n][m] = sum_d q[d,h,n] k[d,h,m] / 8  (channel = d*4+h)
      for (int h = 0; h < HEADS; ++h) {
        launch_gemm(stream,
                    qb16 + (long)h * N, 1, 4L * N, DN,      // A[n][d]
                    kb16 + (long)h * N, 4L * N, 1, DN,      // B[d][m]
                    bigf + (long)h * N * N, nullptr, N, (long)HEADS * N * N,
                    N, N, 64, BATCH, 0.125f, GF_OUTF);
      }
      softmax_rows_bf16_kernel<<<blocks((long)BATCH * HEADS * N * 32), TPB, 0, stream>>>(
          bigf, attnb, N, (long)BATCH * HEADS * N);
      // out[d,h,n] = sum_m attn[h,n,m] v[d,h,m]  (TDM double-buffer path)
      for (int h = 0; h < HEADS; ++h) {
        launch_gemm(stream,
                    vb16 + (long)h * N, 4L * N, 1, DN,      // A[d][m']
                    attnb + (long)h * N * N, 1, N, (long)HEADS * N * N, // B[m'][n]
                    nullptr, ob16 + (long)h * N, 4L * N, DN,
                    64, N, N, BATCH, 1.0f, GF_OUTB);
      }
      // merge -> upper half of concat buffer (rows 256..511)
      launch_gemm(stream, gmw[l], D, 1, 0, ob16, N, 1, DN,
                  nullptr, cc + DN, N, 2 * DN,
                  D, N, D, BATCH, 1.0f, GF_BIAS | GF_OUTB, mb);
    }
    // concat lower halves after both m0,m1 are done
    copy_rows_u16_kernel<<<blocks((long)BATCH * DN), TPB, 0, stream>>>(
        d0b, DN, c0b, 2 * DN, DN, BATCH);
    copy_rows_u16_kernel<<<blocks((long)BATCH * DN), TPB, 0, stream>>>(
        d1b, DN, c1b_, 2 * DN, DN, BATCH);
    for (int t = 0; t < 2; ++t) {
      float* ddf = t ? d1f : d0f;
      u16*   ddb = t ? d1b : d0b;
      u16*   cc  = t ? c1b_ : c0b;
      launch_gemm(stream, gc1w[l], 2 * D, 1, 0, cc, N, 1, 2 * DN,
                  nullptr, hidb, N, 2 * DN,
                  2 * D, N, 2 * D, BATCH, 1.0f, GF_BIAS | GF_BNRELU | GF_OUTB,
                  c1b, bngp, bnbp, bnmp, bnvp);
      launch_gemm(stream, gc2w[l], 2 * D, 1, 0, hidb, N, 1, 2 * DN,
                  ddf, ddb, N, DN,
                  D, N, 2 * D, BATCH, 1.0f, GF_BIAS | GF_RES | GF_OUTF | GF_OUTB,
                  c2b, nullptr, nullptr, nullptr, nullptr, ddf, DN);
    }
  }

  // =====================  final projection + score matrix  =====================
  launch_gemm(stream, fpw, D, 1, 0, d0b, N, 1, DN, nullptr, f0b, N, DN,
              D, N, D, BATCH, 1.0f, GF_BIAS | GF_OUTB, P(FP_B));
  launch_gemm(stream, fpw, D, 1, 0, d1b, N, 1, DN, nullptr, f1b, N, DN,
              D, N, D, BATCH, 1.0f, GF_BIAS | GF_OUTB, P(FP_B));
  float* scmat = bigf;
  float* cpl   = bigf + (long)BATCH * N * N;
  launch_gemm(stream,
              f0b, 1, (long)N, DN,      // A[n][d] = f0[d][n]
              f1b, (long)N, 1, DN,      // B[d][m] = f1[d][m]
              scmat, nullptr, N, (long)N * N,
              N, N, D, BATCH, 0.0625f, GF_OUTF);

  // =====================  Sinkhorn  =====================
  const float norm  = -logf(2048.0f);
  const float logns = logf(1024.0f);
  build_couplings_kernel<<<blocks((long)BATCH * NZ * NZ), TPB, 0, stream>>>(
      scmat, alpha, cpl, N, N, BATCH);
  fill_kernel<<<blocks(2L * BATCH * NZ), TPB, 0, stream>>>(uu, 0.0f, 2L * BATCH * NZ);
  for (int it = 0; it < 20; ++it) {
    sink_row_kernel<<<blocks((long)BATCH * NZ * 32), TPB, 0, stream>>>(
        cpl, vv, uu, NZ, BATCH, norm, logns);
    sink_col_kernel<<<blocks((long)BATCH * NZ), TPB, 0, stream>>>(
        cpl, uu, vv, NZ, BATCH, norm, logns);
  }
  finalize_z_kernel<<<blocks((long)BATCH * NZ * NZ), TPB, 0, stream>>>(
      cpl, uu, vv, Z, NZ, BATCH, norm);

  // =====================  match extraction  =====================
  argmax_row_kernel<<<blocks((long)BATCH * N * 32), TPB, 0, stream>>>(
      Z, mx0, ix0, N, NZ, BATCH);
  argmax_col_kernel<<<blocks((long)BATCH * N), TPB, 0, stream>>>(
      Z, mx1, ix1, N, NZ, BATCH);
  match0_kernel<<<blocks((long)BATCH * N), TPB, 0, stream>>>(
      mx0, ix0, ix1, matches0, msc0, N, BATCH);
  match1_kernel<<<blocks((long)BATCH * N), TPB, 0, stream>>>(
      ix0, ix1, matches0, msc0, matches1, msc1, N, BATCH);
}
